// MoEForEmotionAndTriggerClassification_37288906064211
// MI455X (gfx1250) — matrix operand order repaired and checked by reading, not compiled, not run on hardware
//
#include <hip/hip_runtime.h>
#include <hip/hip_bf16.h>

// ---------------------------------------------------------------------------
// Problem constants (from reference)
// ---------------------------------------------------------------------------
static constexpr int Bb = 128;   // batch
static constexpr int Ss = 512;   // seq
static constexpr int Hh = 768;   // hidden
static constexpr int Ee = 16;    // experts
static constexpr int Kk = 2;     // top-k
static constexpr int Cc = 7;     // emotion classes
static constexpr int Ff = 512;   // ffn dim

// ---------------------------------------------------------------------------
// CDNA5 WMMA types
// ---------------------------------------------------------------------------
typedef __attribute__((ext_vector_type(16))) __bf16 v16bf;
typedef __attribute__((ext_vector_type(8)))  __bf16 v8bf;
typedef __attribute__((ext_vector_type(8)))  float  v8f;

union ABu { v16bf v; v8bf h[2]; };

__device__ __forceinline__ __bf16 f2bf(float f) {
    union { float f; unsigned u; } in; in.f = f;
    unsigned u = in.u;
    unsigned r = (u + 0x7FFFu + ((u >> 16) & 1u)) >> 16;   // round-to-nearest-even
    union { unsigned short s; __bf16 b; } out; out.s = (unsigned short)r;
    return out.b;
}

// A-tile (16x32 bf16, row-major source, row stride `ld` elements).
// Lane L (L%16 = row, L/16 = K phase): elems 0..7 = K k0+phase*8, elems 8..15 = +16.
__device__ __forceinline__ v16bf load_tileA(const __bf16* base) {
    ABu a;
    a.h[0] = *(const v8bf*)(base);
    a.h[1] = *(const v8bf*)(base + 16);
    return a.v;
}
// B-tile (32x16 bf16) from N-major / K-contiguous weights.
// Lane L (L%16 = col, L/16 = K phase): elems 0..7 = K k0+phase*8, elems 8..15 = +16.
__device__ __forceinline__ v16bf load_tileB(const __bf16* base) {
    ABu b;
    b.h[0] = *(const v8bf*)(base);
    b.h[1] = *(const v8bf*)(base + 16);
    return b.v;
}

__device__ __forceinline__ v8f wmma_bf16(v16bf a, v16bf b, v8f c) {
    return __builtin_amdgcn_wmma_f32_16x16x32_bf16(
        /*neg_a=*/false, a, /*neg_b=*/false, b,
        /*c_mod=*/(short)0, c, /*reuse_a=*/false, /*reuse_b=*/false);
}

// ---------------------------------------------------------------------------
// Prep kernels: fp32 -> bf16 conversions / transposes into workspace
// ---------------------------------------------------------------------------
__global__ void cvt_bf16_kernel(const float* __restrict__ src,
                                __bf16* __restrict__ dst, size_t n) {
    size_t i = (size_t)blockIdx.x * blockDim.x + threadIdx.x;
    if (i < n) dst[i] = f2bf(src[i]);
}

// src (E, D1, D2) fp32 -> dst (E, D2, D1) bf16   (N-major, K-contiguous)
__global__ void transpose_e_kernel(const float* __restrict__ src,
                                   __bf16* __restrict__ dst, int D1, int D2) {
    size_t i = (size_t)blockIdx.x * blockDim.x + threadIdx.x;
    size_t tot = (size_t)Ee * D1 * D2;
    if (i >= tot) return;
    size_t per = (size_t)D1 * D2;
    int e = (int)(i / per);
    size_t rem = i % per;
    int j = (int)(rem % D1);   // K index (dim D1)
    int l = (int)(rem / D1);   // N index (dim D2)
    dst[i] = f2bf(src[(size_t)e * per + (size_t)j * D2 + l]);
}

// Head weight, padded to 16 cols: cols 0..6 = We (H,C), col 7 = Wt (H,1), rest 0.
// Stored N-major: WhT[n][h], 16 x 768.
__global__ void build_whead_kernel(const float* __restrict__ We,
                                   const float* __restrict__ Wt,
                                   __bf16* __restrict__ WhT) {
    int i = blockIdx.x * blockDim.x + threadIdx.x;
    if (i >= 16 * Hh) return;
    int n = i / Hh, h = i % Hh;
    float v = (n < Cc) ? We[h * Cc + n] : ((n == 7) ? Wt[h] : 0.0f);
    WhT[i] = f2bf(v);
}

// d_out = [emotion_logits (B,S,C) | trigger (B,S)]; prefill with biases,
// expert contributions are atomically accumulated on top.
__global__ void init_out_kernel(const float* __restrict__ be,
                                const float* __restrict__ bt,
                                float* __restrict__ outp) {
    int i = blockIdx.x * blockDim.x + threadIdx.x;
    const int ne = Bb * Ss * Cc;
    const int nt = Bb * Ss;
    if (i < ne)            outp[i] = be[i % Cc];
    else if (i < ne + nt)  outp[i] = bt[0];
}

// ---------------------------------------------------------------------------
// Gating: mean-pool -> softmax -> top-2. One block per batch row.
// ---------------------------------------------------------------------------
__global__ __launch_bounds__(256)
void gate_kernel(const float* __restrict__ emb, const float* __restrict__ Wg,
                 const float* __restrict__ bg, int* __restrict__ topi,
                 float* __restrict__ topw) {
    __shared__ float pooled[Hh];
    __shared__ float logits[Ee];
    const int b = blockIdx.x, t = threadIdx.x;
    for (int h = t; h < Hh; h += 256) {
        float s = 0.f;
        const float* p = emb + (size_t)b * Ss * Hh + h;
        for (int si = 0; si < Ss; ++si) s += p[(size_t)si * Hh];
        pooled[h] = s * (1.0f / (float)Ss);
    }
    __syncthreads();
    if (t < Ee) {
        float s = bg[t];
        for (int h = 0; h < Hh; ++h) s += pooled[h] * Wg[h * Ee + t];
        logits[t] = s;
    }
    __syncthreads();
    if (t == 0) {
        float mx = -1e30f;
        for (int i = 0; i < Ee; ++i) mx = fmaxf(mx, logits[i]);
        float ex[Ee]; float sum = 0.f;
        for (int i = 0; i < Ee; ++i) { ex[i] = __expf(logits[i] - mx); sum += ex[i]; }
        int i1 = 0;
        for (int i = 1; i < Ee; ++i) if (ex[i] > ex[i1]) i1 = i;
        int i2 = (i1 == 0) ? 1 : 0;
        for (int i = 0; i < Ee; ++i) if (i != i1 && ex[i] > ex[i2]) i2 = i;
        float inv = 1.0f / sum;
        topi[b * Kk + 0] = i1; topw[b * Kk + 0] = ex[i1] * inv;
        topi[b * Kk + 1] = i2; topw[b * Kk + 1] = ex[i2] * inv;
    }
}

// ---------------------------------------------------------------------------
// Fused expert FFN + weighted head contribution.
// Block = 256 threads (8 wave32). Grid = B*K*(S/32).
// Per block: 32 seq rows of one (b, k):
//   GEMM1 (32xH @ HxF) -> relu(+b1) -> LDS bf16
//   GEMM2 (32xF @ FxH) -> w*( +b2 ) -> LDS bf16
//   head  (32xH @ Hx16) -> atomicAdd into d_out (emotion cols 0..6, trigger col 7)
// ---------------------------------------------------------------------------
#define HSTR 520   // 512 + 8 pad halfs
#define OSTR 776   // 768 + 8 pad halfs

__global__ __launch_bounds__(256)
void moe_main_kernel(const __bf16* __restrict__ Xbf,
                     const __bf16* __restrict__ W1T,
                     const __bf16* __restrict__ W2T,
                     const __bf16* __restrict__ WhT,
                     const float* __restrict__ b1p,
                     const float* __restrict__ b2p,
                     const int* __restrict__ topi,
                     const float* __restrict__ topw,
                     float* __restrict__ outp) {
    // h-tile (32 x HSTR) and out-tile (32 x OSTR) share one buffer; barriers
    // separate the read/write phases.
    __shared__ __align__(16) __bf16 smem[32 * OSTR];
    __bf16* hsm = smem;
    __bf16* osm = smem;

    const int bid = blockIdx.x;
    const int b   = bid >> 5;          // K*(S/32) = 32 blocks per batch row
    const int k   = (bid >> 4) & 1;
    const int s0  = (bid & 15) * 32;

    const int   e   = topi[b * Kk + k];
    const float wgt = topw[b * Kk + k];

    const int tid   = threadIdx.x;
    const int wave  = tid >> 5;
    const int lane  = tid & 31;
    const int lhalf = lane >> 4;       // K phase
    const int lrow  = lane & 15;       // A row / B col / C col

    // ---------------- GEMM1: h = relu(X @ W1 + b1) ----------------
    {
        v8f acc[2][4];
        const v8f zacc = {0.f,0.f,0.f,0.f,0.f,0.f,0.f,0.f};
        #pragma unroll
        for (int m = 0; m < 2; ++m)
            #pragma unroll
            for (int n = 0; n < 4; ++n) acc[m][n] = zacc;

        const int n0 = wave * 64;                       // F columns per wave
        const __bf16* xbase = Xbf + (size_t)b * Ss * Hh + (size_t)s0 * Hh;
        const __bf16* wbase = W1T + (size_t)e * Ff * Hh;

        for (int k0 = 0; k0 < Hh; k0 += 32) {
            v16bf a[2], bt[4];
            #pragma unroll
            for (int m = 0; m < 2; ++m)
                a[m] = load_tileA(xbase + (size_t)(m * 16 + lrow) * Hh + k0 + lhalf * 8);
            #pragma unroll
            for (int n = 0; n < 4; ++n)
                bt[n] = load_tileB(wbase + (size_t)(n0 + n * 16 + lrow) * Hh + k0 + lhalf * 8);
            #pragma unroll
            for (int m = 0; m < 2; ++m)
                #pragma unroll
                for (int n = 0; n < 4; ++n)
                    acc[m][n] = wmma_bf16(a[m], bt[n], acc[m][n]);
        }
        #pragma unroll
        for (int m = 0; m < 2; ++m)
            #pragma unroll
            for (int n = 0; n < 4; ++n) {
                const int col = n0 + n * 16 + lrow;
                const float bias = b1p[e * Ff + col];
                #pragma unroll
                for (int r = 0; r < 8; ++r) {
                    const int row = m * 16 + lhalf * 8 + r;
                    float v = acc[m][n][r] + bias;
                    hsm[row * HSTR + col] = f2bf(v > 0.f ? v : 0.f);
                }
            }
    }
    __syncthreads();

    // ---------------- GEMM2: out = wgt * (h @ W2 + b2) ----------------
    v8f acc2[2][6];
    {
        const v8f zacc = {0.f,0.f,0.f,0.f,0.f,0.f,0.f,0.f};
        #pragma unroll
        for (int m = 0; m < 2; ++m)
            #pragma unroll
            for (int n = 0; n < 6; ++n) acc2[m][n] = zacc;

        const int n0 = wave * 96;                       // H columns per wave
        const __bf16* wbase = W2T + (size_t)e * Hh * Ff;

        for (int k0 = 0; k0 < Ff; k0 += 32) {
            v16bf a[2], bt[6];
            #pragma unroll
            for (int m = 0; m < 2; ++m)
                a[m] = load_tileA(&hsm[(m * 16 + lrow) * HSTR + k0 + lhalf * 8]);
            #pragma unroll
            for (int n = 0; n < 6; ++n)
                bt[n] = load_tileB(wbase + (size_t)(n0 + n * 16 + lrow) * Ff + k0 + lhalf * 8);
            #pragma unroll
            for (int m = 0; m < 2; ++m)
                #pragma unroll
                for (int n = 0; n < 6; ++n)
                    acc2[m][n] = wmma_bf16(a[m], bt[n], acc2[m][n]);
        }
    }
    __syncthreads();   // all waves finished reading hsm; now reuse buffer as osm
    {
        const int n0 = wave * 96;
        #pragma unroll
        for (int m = 0; m < 2; ++m)
            #pragma unroll
            for (int n = 0; n < 6; ++n) {
                const int col = n0 + n * 16 + lrow;
                const float bias = b2p[e * Hh + col];
                #pragma unroll
                for (int r = 0; r < 8; ++r) {
                    const int row = m * 16 + lhalf * 8 + r;
                    osm[row * OSTR + col] = f2bf(wgt * (acc2[m][n][r] + bias));
                }
            }
    }
    __syncthreads();

    // ---------------- Head: (32 x 768) @ (768 x 16) on waves 0-1 ----------------
    if (wave < 2) {
        v8f acc = {0.f,0.f,0.f,0.f,0.f,0.f,0.f,0.f};
        for (int k0 = 0; k0 < Hh; k0 += 32) {
            v16bf a  = load_tileA(&osm[(wave * 16 + lrow) * OSTR + k0 + lhalf * 8]);
            v16bf bb = load_tileB(WhT + (size_t)lrow * Hh + k0 + lhalf * 8);
            acc = wmma_bf16(a, bb, acc);
        }
        float* emo = outp;
        float* trg = outp + (size_t)Bb * Ss * Cc;
        #pragma unroll
        for (int r = 0; r < 8; ++r) {
            const int srow = s0 + wave * 16 + lhalf * 8 + r;
            const float v = acc[r];
            if (lrow < Cc)
                atomicAdd(&emo[((size_t)b * Ss + srow) * Cc + lrow], v);
            else if (lrow == 7)
                atomicAdd(&trg[(size_t)b * Ss + srow], v);
        }
    }
}

// ---------------------------------------------------------------------------
// Host launcher
// ---------------------------------------------------------------------------
extern "C" void kernel_launch(void* const* d_in, const int* in_sizes, int n_in,
                              void* d_out, int out_size, void* d_ws, size_t ws_size,
                              hipStream_t stream) {
    const float* emb = (const float*)d_in[0];
    const float* Wg  = (const float*)d_in[1];
    const float* bg  = (const float*)d_in[2];
    const float* W1  = (const float*)d_in[3];
    const float* b1  = (const float*)d_in[4];
    const float* W2  = (const float*)d_in[5];
    const float* b2  = (const float*)d_in[6];
    const float* We  = (const float*)d_in[7];
    const float* be  = (const float*)d_in[8];
    const float* Wt  = (const float*)d_in[9];
    const float* bt  = (const float*)d_in[10];
    float* outp = (float*)d_out;

    // workspace layout
    char* ws = (char*)d_ws;
    int*    topi = (int*)(ws + 0);                       // B*K ints
    float*  topw = (float*)(ws + 1024);                  // B*K floats
    __bf16* WhT  = (__bf16*)(ws + 4096);                 // 16*768 bf16
    __bf16* W1T  = (__bf16*)(ws + 32768);                // E*F*H bf16 (N-major)
    const size_t wsz = (size_t)Ee * Ff * Hh;
    __bf16* W2T  = W1T + wsz;                            // E*H*F bf16 (N-major)
    __bf16* Xbf  = W2T + wsz;                            // B*S*H bf16
    const size_t xn = (size_t)Bb * Ss * Hh;

    // prep: bf16 weights (transposed to N-major/K-contiguous) + bf16 activations
    transpose_e_kernel<<<(unsigned)((wsz + 255) / 256), 256, 0, stream>>>(W1, W1T, Hh, Ff);
    transpose_e_kernel<<<(unsigned)((wsz + 255) / 256), 256, 0, stream>>>(W2, W2T, Ff, Hh);
    build_whead_kernel<<<(16 * Hh + 255) / 256, 256, 0, stream>>>(We, Wt, WhT);
    cvt_bf16_kernel<<<(unsigned)((xn + 255) / 256), 256, 0, stream>>>(emb, Xbf, xn);

    // prefill output with head biases (contributions accumulate atomically)
    init_out_kernel<<<(Bb * Ss * (Cc + 1) + 255) / 256, 256, 0, stream>>>(be, bt, outp);

    // gating
    gate_kernel<<<Bb, 256, 0, stream>>>(emb, Wg, bg, topi, topw);

    // fused MoE FFN + heads: B*K*(S/32) = 4096 workgroups
    moe_main_kernel<<<Bb * Kk * (Ss / 32), 256, 0, stream>>>(
        Xbf, W1T, W2T, WhT, b1, b2, topi, topw, outp);
}
